// GAT_43198781063769
// MI455X (gfx1250) — matrix operand order repaired and checked by reading, not compiled, not run on hardware
//
#include <hip/hip_runtime.h>
#include <hip/hip_bf16.h>

typedef __attribute__((ext_vector_type(16))) __bf16 v16bf;
typedef __attribute__((ext_vector_type(8)))  __bf16 v8bf;
typedef __attribute__((ext_vector_type(4)))  __bf16 v4bf;
typedef __attribute__((ext_vector_type(8)))  float  v8f;

#define N_NODES   100000
#define N_EDGES   1600000
#define E_TOT     (N_EDGES + N_NODES)
#define IN_DIM    128
#define HID       64
#define N_CLASSES 40
#define NEG_SLOPE 0.2f

// ---------- helpers ----------
static __device__ __forceinline__ unsigned short bf16_bits(float f) {
    unsigned u = __float_as_uint(f);
    u += 0x7FFFu + ((u >> 16) & 1u);          // round-to-nearest-even
    return (unsigned short)(u >> 16);
}
static __device__ __forceinline__ __bf16 bf16_of(float f) {
    unsigned short h = bf16_bits(f);
    return __builtin_bit_cast(__bf16, h);
}
// order-preserving float<->uint encoding so atomicMax(u32) == float max
static __device__ __forceinline__ unsigned ford(float f) {
    unsigned u = __float_as_uint(f);
    return (u & 0x80000000u) ? ~u : (u | 0x80000000u);
}
static __device__ __forceinline__ float fdec(unsigned u) {
    unsigned v = (u & 0x80000000u) ? (u & 0x7FFFFFFFu) : ~u;
    return __uint_as_float(v);
}

// ---------- f32 -> bf16 bulk convert (float4 in, 4x bf16 out) ----------
__global__ __launch_bounds__(256) void cvt_f32_bf16(const float* __restrict__ in,
                                                    __bf16* __restrict__ out, int n4) {
    int i = blockIdx.x * blockDim.x + threadIdx.x;
    if (i >= n4) return;
    float4 f = ((const float4*)in)[i];
    v4bf o;
    o[0] = bf16_of(f.x); o[1] = bf16_of(f.y); o[2] = bf16_of(f.z); o[3] = bf16_of(f.w);
    ((v4bf*)out)[i] = o;
}

// ---------- GEMM: C[N x 64] = A[N x K](bf16) @ W[K x 64](f32), bf16 WMMA ----
// One wave owns a 16-row tile and all four 16-col tiles: A-frag loaded once,
// 4 WMMAs per K-step.
template <int K>
__global__ __launch_bounds__(256) void gemm_bf16_wmma(const __bf16* __restrict__ A,
                                                      const float* __restrict__ W,
                                                      float* __restrict__ C) {
    __shared__ __bf16 wl[64 * K];              // col-major: wl[n*K + k]
    const int tid = threadIdx.x;
    for (int idx = tid; idx < 64 * K; idx += 256) {
        int n = idx / K, k = idx % K;
        wl[idx] = bf16_of(W[k * 64 + n]);
    }
    __syncthreads();

    const int lane = tid & 31;
    const int wave = tid >> 5;
    const int m0   = (blockIdx.x * 8 + wave) * 16;   // 16-row tile per wave
    if (m0 >= N_NODES) return;                       // uniform per-wave guard
    const int m    = lane & 15;
    const int half = lane >> 4;

    v8f c0 = {}, c1 = {}, c2 = {}, c3 = {};
    for (int k0 = 0; k0 < K; k0 += 32) {
        // A fragment: elems 0-15 = K[half*16..half*16+15]
        const __bf16* arow = A + (size_t)(m0 + m) * K + k0 + half * 16;
        v8bf lo = *(const v8bf*)arow;              // global_load_b128
        v8bf hi = *(const v8bf*)(arow + 8);        // global_load_b128
        v16bf a = __builtin_shufflevector(lo, hi, 0,1,2,3,4,5,6,7,8,9,10,11,12,13,14,15);
        // B fragments: 16 contiguous bf16 per col in LDS -> ds_load_b128 x2
        const int kb = k0 + half * 16;
        v16bf b0 = *(const v16bf*)&wl[(0  + m) * K + kb];
        v16bf b1 = *(const v16bf*)&wl[(16 + m) * K + kb];
        v16bf b2 = *(const v16bf*)&wl[(32 + m) * K + kb];
        v16bf b3 = *(const v16bf*)&wl[(48 + m) * K + kb];
        c0 = __builtin_amdgcn_wmma_f32_16x16x32_bf16(false, a, false, b0, (short)0, c0, false, false);
        c1 = __builtin_amdgcn_wmma_f32_16x16x32_bf16(false, a, false, b1, (short)0, c1, false, false);
        c2 = __builtin_amdgcn_wmma_f32_16x16x32_bf16(false, a, false, b2, (short)0, c2, false, false);
        c3 = __builtin_amdgcn_wmma_f32_16x16x32_bf16(false, a, false, b3, (short)0, c3, false, false);
    }
    float* crow = C + (size_t)(m0 + half * 8) * 64 + m;
#pragma unroll
    for (int i = 0; i < 8; ++i) {
        crow[(size_t)i * 64 + 0]  = c0[i];
        crow[(size_t)i * 64 + 16] = c1[i];
        crow[(size_t)i * 64 + 32] = c2[i];
        crow[(size_t)i * 64 + 48] = c3[i];
    }
}

// ---------- per-node: s = h.a_s, d = h.a_d; zero m/denom/acc ----------
__global__ __launch_bounds__(256) void node_prep(const float* __restrict__ h,
                                                 const float* __restrict__ a_s,
                                                 const float* __restrict__ a_d,
                                                 float* __restrict__ s, float* __restrict__ d,
                                                 unsigned* __restrict__ menc,
                                                 float* __restrict__ denom,
                                                 float* __restrict__ acc) {
    int node = (blockIdx.x * blockDim.x + threadIdx.x) >> 5;
    int lane = threadIdx.x & 31;
    if (node >= N_NODES) return;
    const float* hr = h + (size_t)node * HID;
    float h0 = hr[lane], h1 = hr[lane + 32];
    float sv = h0 * a_s[lane] + h1 * a_s[lane + 32];
    float dv = h0 * a_d[lane] + h1 * a_d[lane + 32];
#pragma unroll
    for (int off = 16; off; off >>= 1) {
        sv += __shfl_xor(sv, off, 32);
        dv += __shfl_xor(dv, off, 32);
    }
    acc[(size_t)node * HID + lane]      = 0.f;
    acc[(size_t)node * HID + lane + 32] = 0.f;
    if (lane == 0) { s[node] = sv; d[node] = dv; menc[node] = 0u; denom[node] = 0.f; }
}

// ---------- edge pass 1: e = leaky_relu(s[src]+d[dst]); segment max ----------
__global__ __launch_bounds__(256) void edge_max(const long long* __restrict__ ei,
                                                const float* __restrict__ s,
                                                const float* __restrict__ d,
                                                float* __restrict__ e_buf,
                                                unsigned* __restrict__ menc) {
    int t = blockIdx.x * blockDim.x + threadIdx.x;
    if (t >= E_TOT) return;
    int src, dst;
    if (t < N_EDGES) { src = (int)ei[t]; dst = (int)ei[N_EDGES + t]; }
    else             { src = dst = t - N_EDGES; }
    float e = s[src] + d[dst];
    e = e > 0.f ? e : NEG_SLOPE * e;
    e_buf[t] = e;
    atomicMax(&menc[dst], ford(e));
}

// ---------- edge pass 2: p = exp(e-m); denom += p; acc[dst] += p*h[src] ------
__global__ __launch_bounds__(256) void edge_accum(const long long* __restrict__ ei,
                                                  const float* __restrict__ e_buf,
                                                  const unsigned* __restrict__ menc,
                                                  float* __restrict__ denom,
                                                  const float* __restrict__ h,
                                                  float* __restrict__ acc) {
    int wv   = (blockIdx.x * blockDim.x + threadIdx.x) >> 5;  // one wave32 per edge
    int lane = threadIdx.x & 31;
    if (wv >= E_TOT) return;
    int src, dst;
    if (wv < N_EDGES) { src = (int)ei[wv]; dst = (int)ei[N_EDGES + wv]; }
    else              { src = dst = wv - N_EDGES; }
    float p = __expf(e_buf[wv] - fdec(menc[dst]));
    if (lane == 0) atomicAdd(&denom[dst], p);
    const float* hr = h + (size_t)src * HID;
    float* ar = acc + (size_t)dst * HID;
    atomicAdd(&ar[lane],      p * hr[lane]);
    atomicAdd(&ar[lane + 32], p * hr[lane + 32]);
}

// ---------- per-node finalize: normalize + bias + ELU -> bf16 ----------
__global__ __launch_bounds__(256) void node_finish(const float* __restrict__ acc,
                                                   const float* __restrict__ denom,
                                                   const float* __restrict__ b,
                                                   __bf16* __restrict__ out) {
    int idx = blockIdx.x * blockDim.x + threadIdx.x;
    if (idx >= N_NODES * HID) return;
    int node = idx >> 6, k = idx & 63;
    float v = acc[idx] / (denom[node] + 1e-16f) + b[k];
    v = v > 0.f ? v : (__expf(v) - 1.f);
    out[idx] = bf16_of(v);
}

// ---------- head: out[N x 40] = A[N x 64](bf16) @ Wl[64 x 40] + bl ----------
__global__ __launch_bounds__(256) void gemm_out_wmma(const __bf16* __restrict__ A,
                                                     const float* __restrict__ Wl,
                                                     const float* __restrict__ bl,
                                                     float* __restrict__ out) {
    __shared__ __bf16 wl[48 * 64];             // col-major, padded to 48 cols
    const int tid = threadIdx.x;
    for (int idx = tid; idx < 48 * 64; idx += 256) {
        int n = idx / 64, k = idx % 64;
        wl[idx] = (n < N_CLASSES) ? bf16_of(Wl[k * N_CLASSES + n]) : bf16_of(0.f);
    }
    __syncthreads();

    const int lane = tid & 31;
    const int wave = tid >> 5;
    const int m0   = (blockIdx.x * 8 + wave) * 16;
    if (m0 >= N_NODES) return;
    const int m    = lane & 15;
    const int half = lane >> 4;

    v8f c0 = {}, c1 = {}, c2 = {};
    for (int k0 = 0; k0 < 64; k0 += 32) {
        const __bf16* arow = A + (size_t)(m0 + m) * 64 + k0 + half * 16;
        v8bf lo = *(const v8bf*)arow;
        v8bf hi = *(const v8bf*)(arow + 8);
        v16bf a = __builtin_shufflevector(lo, hi, 0,1,2,3,4,5,6,7,8,9,10,11,12,13,14,15);
        const int kb = k0 + half * 16;
        v16bf b0 = *(const v16bf*)&wl[(0  + m) * 64 + kb];
        v16bf b1 = *(const v16bf*)&wl[(16 + m) * 64 + kb];
        v16bf b2 = *(const v16bf*)&wl[(32 + m) * 64 + kb];
        c0 = __builtin_amdgcn_wmma_f32_16x16x32_bf16(false, a, false, b0, (short)0, c0, false, false);
        c1 = __builtin_amdgcn_wmma_f32_16x16x32_bf16(false, a, false, b1, (short)0, c1, false, false);
        c2 = __builtin_amdgcn_wmma_f32_16x16x32_bf16(false, a, false, b2, (short)0, c2, false, false);
    }
#pragma unroll
    for (int t = 0; t < 3; ++t) {
        int n = t * 16 + m;
        if (n < N_CLASSES) {
            float bb = bl[n];
            v8f c = (t == 0) ? c0 : (t == 1) ? c1 : c2;
#pragma unroll
            for (int i = 0; i < 8; ++i)
                out[(size_t)(m0 + half * 8 + i) * N_CLASSES + n] = c[i] + bb;
        }
    }
}

// ---------- host launcher ----------
extern "C" void kernel_launch(void* const* d_in, const int* in_sizes, int n_in,
                              void* d_out, int out_size, void* d_ws, size_t ws_size,
                              hipStream_t stream) {
    (void)in_sizes; (void)n_in; (void)out_size; (void)ws_size;
    const float*     x    = (const float*)d_in[0];
    const long long* ei   = (const long long*)d_in[1];   // int64 [2, E]
    const float*     W0   = (const float*)d_in[2];
    const float*     a_s0 = (const float*)d_in[3];
    const float*     a_d0 = (const float*)d_in[4];
    const float*     b0   = (const float*)d_in[5];
    const float*     W1   = (const float*)d_in[6];
    const float*     a_s1 = (const float*)d_in[7];
    const float*     a_d1 = (const float*)d_in[8];
    const float*     b1   = (const float*)d_in[9];
    const float*     Wl   = (const float*)d_in[10];
    const float*     bl   = (const float*)d_in[11];
    float* outp = (float*)d_out;

    // workspace layout
    __bf16*   xb    = (__bf16*)d_ws;                          // N*128 bf16
    __bf16*   hinb  = xb + (size_t)N_NODES * IN_DIM;          // N*64  bf16
    float*    h     = (float*)(hinb + (size_t)N_NODES * HID); // N*64  f32
    float*    acc   = h + (size_t)N_NODES * HID;              // N*64  f32
    float*    sbuf  = acc + (size_t)N_NODES * HID;            // N
    float*    dbuf  = sbuf + N_NODES;                         // N
    unsigned* menc  = (unsigned*)(dbuf + N_NODES);            // N
    float*    denom = (float*)(menc + N_NODES);               // N
    float*    e_buf = denom + N_NODES;                        // E_TOT

    const dim3 B256(256);
    const dim3 Gcvt((N_NODES * IN_DIM / 4 + 255) / 256);
    const dim3 Ggemm((N_NODES / 16 + 7) / 8);       // 16-row tile per wave, 8 waves/block
    const dim3 Gnode(N_NODES / 8);                  // wave per node
    const dim3 Gemax((E_TOT + 255) / 256);
    const dim3 Geacc(E_TOT / 8);                    // wave per edge
    const dim3 Gfin((N_NODES * HID) / 256);

    // ---- layer 0 ----
    cvt_f32_bf16<<<Gcvt, B256, 0, stream>>>(x, xb, N_NODES * IN_DIM / 4);
    gemm_bf16_wmma<IN_DIM><<<Ggemm, B256, 0, stream>>>(xb, W0, h);
    node_prep<<<Gnode, B256, 0, stream>>>(h, a_s0, a_d0, sbuf, dbuf, menc, denom, acc);
    edge_max<<<Gemax, B256, 0, stream>>>(ei, sbuf, dbuf, e_buf, menc);
    edge_accum<<<Geacc, B256, 0, stream>>>(ei, e_buf, menc, denom, h, acc);
    node_finish<<<Gfin, B256, 0, stream>>>(acc, denom, b0, hinb);

    // ---- layer 1 ----
    gemm_bf16_wmma<HID><<<Ggemm, B256, 0, stream>>>(hinb, W1, h);
    node_prep<<<Gnode, B256, 0, stream>>>(h, a_s1, a_d1, sbuf, dbuf, menc, denom, acc);
    edge_max<<<Gemax, B256, 0, stream>>>(ei, sbuf, dbuf, e_buf, menc);
    edge_accum<<<Geacc, B256, 0, stream>>>(ei, e_buf, menc, denom, h, acc);
    node_finish<<<Gfin, B256, 0, stream>>>(acc, denom, b1, hinb);

    // ---- head ----
    gemm_out_wmma<<<Ggemm, B256, 0, stream>>>(hinb, Wl, bl, outp);
}
